// MultiHeadAttention_27882927685681
// MI455X (gfx1250) — matrix operand order repaired and checked by reading, not compiled, not run on hardware
//
#include <hip/hip_runtime.h>
#include <hip/hip_bf16.h>

// Problem constants (B=4, S=2048, E=1024, H=16, D=64), SCALE = sqrt(64) = 8.
#define BB 4
#define SS 2048
#define EE 1024
#define HH 16
#define DD 64
#define INV_SCALE 0.125f

typedef _Float16 f16;
typedef __attribute__((ext_vector_type(16))) _Float16 v16h;
typedef __attribute__((ext_vector_type(8)))  _Float16 v8h;
typedef __attribute__((ext_vector_type(4)))  _Float16 v4h;
typedef __attribute__((ext_vector_type(8)))  float    v8f;
typedef __attribute__((ext_vector_type(4)))  float    v4f;

#if __has_builtin(__builtin_amdgcn_tensor_load_to_lds) && \
    __has_builtin(__builtin_amdgcn_s_wait_tensorcnt)
#define HAVE_TDM 1
#else
#define HAVE_TDM 0
#endif

// ---------------------------------------------------------------------------
// Fragment loaders (CDNA5 wave32 WMMA 16x16x32 f16 layouts, ISA 05_wmma §7.12.2)
//
// A (16x32, MxK), row-major source with contiguous K:
//   lane l: M = l&15 ; lanes 0-15 hold K {0..7, 16..23}, lanes 16-31 hold
//   K {8..15, 24..31}  -> two contiguous 8-element (16B) runs per lane.
__device__ __forceinline__ v16h load_a_f16(const f16* __restrict__ p, int ld,
                                           int row0, int k0, int lane) {
  const int m  = row0 + (lane & 15);
  const int ka = k0 + ((lane >> 4) << 3);
  const f16* r = p + (size_t)m * ld + ka;
  v8h lo = *(const v8h*)(r);
  v8h hi = *(const v8h*)(r + 16);
  v16h a;
#pragma unroll
  for (int i = 0; i < 8; ++i) { a[i] = lo[i]; a[i + 8] = hi[i]; }
  return a;
}

// Same A fragment but fp32 source (attention weights) -> convert on the fly.
__device__ __forceinline__ v16h load_a_f32(const float* __restrict__ p, int ld,
                                           int row0, int k0, int lane) {
  const int m  = row0 + (lane & 15);
  const int ka = k0 + ((lane >> 4) << 3);
  const float* r = p + (size_t)m * ld + ka;
  v4f x0 = *(const v4f*)(r);
  v4f x1 = *(const v4f*)(r + 4);
  v4f x2 = *(const v4f*)(r + 16);
  v4f x3 = *(const v4f*)(r + 20);
  v16h a;
#pragma unroll
  for (int i = 0; i < 4; ++i) {
    a[i]      = (f16)x0[i];
    a[i + 4]  = (f16)x1[i];
    a[i + 8]  = (f16)x2[i];
    a[i + 12] = (f16)x3[i];
  }
  return a;
}

// B (32x16, KxN): source element (k,n) = src[n*ld + k]. lane l: N = l&15;
// lanes 0-15 hold K {0..7, 16..23}, lanes 16-31 hold K {8..15, 24..31}.
__device__ __forceinline__ v16h load_b_f16(const f16* __restrict__ p, int ld,
                                           int n0, int k0, int lane) {
  const int n  = n0 + (lane & 15);
  const int kb = k0 + ((lane >> 4) << 3);
  const f16* r = p + (size_t)n * ld + kb;
  v8h lo = *(const v8h*)(r);
  v8h hi = *(const v8h*)(r + 16);
  v16h b;
#pragma unroll
  for (int i = 0; i < 8; ++i) { b[i] = lo[i]; b[i + 8] = hi[i]; }
  return b;
}

// Same B fragment from an LDS-staged slab laid out [64 rows][32 f16].
__device__ __forceinline__ v16h load_b_lds(const f16* sh, int n0, int lane) {
  const int n  = n0 + (lane & 15);
  const int kb = (lane >> 4) << 3;
  const f16* r = sh + n * 32 + kb;
  v8h lo = *(const v8h*)(r);
  v8h hi = *(const v8h*)(r + 16);
  v16h b;
#pragma unroll
  for (int i = 0; i < 8; ++i) { b[i] = lo[i]; b[i + 8] = hi[i]; }
  return b;
}

#define WMMA_F16(a, b, c) \
  __builtin_amdgcn_wmma_f32_16x16x32_f16(false, (a), false, (b), (short)0, (c), false, false)

// ---------------------------------------------------------------------------
// Tensor Data Mover: 2-D tile load (64 rows x 32 f16) into LDS.
// D# per ISA 08_async_tensor §8.3/8.4: group0 = {count=1, lds_addr,
// global_addr, type=2}; group1 = {data_size=1(2B), tensor_dim0/1,
// tile_dim0=32, tile_dim1=64, tensor_dim0_stride}; groups 2/3 zero (2-D).
// Flat LDS aperture keeps the workgroup-relative byte offset in the low
// 32 bits (ISA §10.2), so truncating the generic shared pointer is the
// descriptor's lds_addr.
__device__ __forceinline__ unsigned lds_off(const void* p) {
  return (unsigned)(unsigned long long)(size_t)p;
}

#if HAVE_TDM
typedef __attribute__((ext_vector_type(4))) unsigned int u32x4;
typedef __attribute__((ext_vector_type(8))) int          i32x8;
typedef __attribute__((ext_vector_type(4))) int          i32x4;

__device__ __forceinline__ void tdm_load_64x32(const f16* gsrc, unsigned lds_byte,
                                               unsigned dim0, unsigned dim1,
                                               unsigned stride0) {
  const unsigned long long ga = (unsigned long long)(size_t)(const void*)gsrc;
  u32x4 g0;
  g0[0] = 1u;                                               // count=1 (user D#)
  g0[1] = lds_byte;                                         // lds_addr (bytes)
  g0[2] = (unsigned)ga;                                     // global_addr lo32
  g0[3] = (unsigned)((ga >> 32) & 0x01FFFFFFu) | 0x80000000u; // addr[56:32] | type=2
  i32x8 g1;
  g1[0] = (int)(1u << 16);                                  // data_size=1 (2 bytes)
  g1[1] = (int)((dim0 & 0xFFFFu) << 16);                    // tensor_dim0 lo16
  g1[2] = (int)((dim0 >> 16) | ((dim1 & 0xFFFFu) << 16));   // dim0 hi | dim1 lo
  g1[3] = (int)((dim1 >> 16) | (32u << 16));                // dim1 hi | tile_dim0=32
  g1[4] = 64;                                               // tile_dim1=64, tile_dim2=0
  g1[5] = (int)stride0;                                     // tensor_dim0_stride lo32
  g1[6] = 0;
  g1[7] = 0;
  const i32x4 z4 = {0, 0, 0, 0};
#if defined(__clang_major__) && (__clang_major__ >= 23)
  const i32x8 z8 = {0, 0, 0, 0, 0, 0, 0, 0};
  __builtin_amdgcn_tensor_load_to_lds(g0, g1, z4, z4, z8, 0);
#else
  __builtin_amdgcn_tensor_load_to_lds(g0, g1, z4, z4, 0);
#endif
}
#endif

// ---------------------------------------------------------------------------
// fp32 -> f16 cast, 4 elements/thread
__global__ __launch_bounds__(256) void cast4_kernel(const float* __restrict__ in,
                                                    f16* __restrict__ out, int n4) {
  int i = blockIdx.x * 256 + threadIdx.x;
  if (i < n4) {
    v4f x = ((const v4f*)in)[i];
    v4h y;
#pragma unroll
    for (int j = 0; j < 4; ++j) y[j] = (f16)x[j];
    ((v4h*)out)[i] = y;
  }
}

// ---------------------------------------------------------------------------
// Fused QKV projection: Y = x @ W.T + b for W in {Wq,Wk,Wv} (blockIdx.y).
// 8 waves/block compute a 128x64 macro-tile sharing one weight slab: the
// 64x32 B-slab streams global->LDS via TDM (double-buffered), fragments are
// read back with ds loads. Q,K stored [B,H,S,D]; V stored [B,H,D,S].
__global__ __launch_bounds__(256) void qkv_proj_kernel(
    const f16* __restrict__ xh,
    const f16* __restrict__ Wqh, const f16* __restrict__ Wkh, const f16* __restrict__ Wvh,
    const float* __restrict__ bq, const float* __restrict__ bk, const float* __restrict__ bv,
    f16* __restrict__ Qh, f16* __restrict__ Kh, f16* __restrict__ Vth) {
  __shared__ __align__(16) f16 shB[2][64 * 32];
  const int lane  = threadIdx.x & 31;
  const int wave  = threadIdx.x >> 5;
  const int which = blockIdx.y;  // 0=Q 1=K 2=V (block-uniform)
  const f16*   W    = (which == 0) ? Wqh : ((which == 1) ? Wkh : Wvh);
  const float* bias = (which == 0) ? bq  : ((which == 1) ? bk  : bv);

  const int nquad = blockIdx.x & 15;                     // shared by the block
  const int m0    = (((int)blockIdx.x >> 4) * 8 + wave) * 16;
  const int n0    = nquad * 64;
  const f16* Bsrc = W + (size_t)n0 * EE;                 // 64 rows x EE

  v8f acc[4] = {v8f{}, v8f{}, v8f{}, v8f{}};
#if HAVE_TDM
  if (wave == 0) {
    tdm_load_64x32(Bsrc, lds_off(&shB[0][0]), EE, 64, EE);
    __builtin_amdgcn_s_wait_tensorcnt(0);
  }
  __syncthreads();
  for (int kk = 0; kk < EE / 32; ++kk) {
    const int buf = kk & 1;
    if (wave == 0 && kk + 1 < EE / 32)
      tdm_load_64x32(Bsrc + (kk + 1) * 32, lds_off(&shB[buf ^ 1][0]), EE, 64, EE);
    v16h a = load_a_f16(xh, EE, m0, kk * 32, lane);
#pragma unroll
    for (int j = 0; j < 4; ++j) {
      v16h b = load_b_lds(&shB[buf][0], j * 16, lane);
      acc[j] = WMMA_F16(a, b, acc[j]);
    }
    if (wave == 0 && kk + 1 < EE / 32) __builtin_amdgcn_s_wait_tensorcnt(0);
    __syncthreads();
  }
#else
  for (int kk = 0; kk < EE / 32; ++kk) {
    for (int idx = threadIdx.x; idx < 64 * 32; idx += 256)
      shB[0][idx] = Bsrc[(size_t)(idx >> 5) * EE + kk * 32 + (idx & 31)];
    __syncthreads();
    v16h a = load_a_f16(xh, EE, m0, kk * 32, lane);
#pragma unroll
    for (int j = 0; j < 4; ++j) {
      v16h b = load_b_lds(&shB[0][0], j * 16, lane);
      acc[j] = WMMA_F16(a, b, acc[j]);
    }
    __syncthreads();
  }
#endif

  const int mrow = m0 + ((lane >> 4) << 3);  // C/D layout: M = r + 8*(lane>=16)
  const int ncol = lane & 15;
#pragma unroll
  for (int j = 0; j < 4; ++j) {
    const int n = n0 + j * 16 + ncol;
    const float bn = bias[n];
    const int h = n >> 6, d = n & 63;
#pragma unroll
    for (int r = 0; r < 8; ++r) {
      const int m = mrow + r;
      const int b = m >> 11, s = m & (SS - 1);
      const f16 val = (f16)(acc[j][r] + bn);
      if (which == 2) {
        Vth[(((size_t)(b * HH + h)) * DD + d) * SS + s] = val;   // [B,H,D,S]
      } else {
        f16* dst = (which == 0) ? Qh : Kh;
        dst[(((size_t)(b * HH + h)) * SS + s) * DD + d] = val;   // [B,H,S,D]
      }
    }
  }
}

// ---------------------------------------------------------------------------
// scores = (Q @ K^T) / 8, fp32, written straight into d_out attn region.
// Wave job = 16x64 tile of one (b,h) S x S score matrix; K-dim = D = 64.
__global__ __launch_bounds__(256) void scores_kernel(
    const f16* __restrict__ Qh, const f16* __restrict__ Kh, float* __restrict__ attn) {
  const int lane = threadIdx.x & 31;
  const int wave = threadIdx.x >> 5;
  const int job  = blockIdx.x * 8 + wave;  // 262144 jobs: 64 bh x 128 mtile x 32 nquad
  const int nquad = job & 31;
  const int mtile = (job >> 5) & 127;
  const int bh    = job >> 12;
  const f16* Q = Qh + (size_t)bh * SS * DD;
  const f16* K = Kh + (size_t)bh * SS * DD;
  const int m0 = mtile * 16, n0 = nquad * 64;

  v8f acc[4] = {v8f{}, v8f{}, v8f{}, v8f{}};
#pragma unroll
  for (int k0 = 0; k0 < DD; k0 += 32) {
    v16h a = load_a_f16(Q, DD, m0, k0, lane);
#pragma unroll
    for (int j = 0; j < 4; ++j) {
      v16h b = load_b_f16(K, DD, n0 + j * 16, k0, lane);
      acc[j] = WMMA_F16(a, b, acc[j]);
    }
  }

  float* A = attn + (size_t)bh * SS * SS;
  const int mrow = m0 + ((lane >> 4) << 3);
  const int ncol = lane & 15;
#pragma unroll
  for (int j = 0; j < 4; ++j) {
    const int n = n0 + j * 16 + ncol;
#pragma unroll
    for (int r = 0; r < 8; ++r) {
      A[(size_t)(mrow + r) * SS + n] = acc[j][r] * INV_SCALE;
    }
  }
}

// ---------------------------------------------------------------------------
// Row softmax in place over the S=2048-wide rows. One 256-thread block / row.
__global__ __launch_bounds__(256) void softmax_kernel(float* __restrict__ attn) {
  __shared__ float red[256];
  float* p = attn + (size_t)blockIdx.x * SS;
  const int t = threadIdx.x;

  float v[8];
  float mx = -3.0e38f;
#pragma unroll
  for (int i = 0; i < 8; ++i) { v[i] = p[t + i * 256]; mx = fmaxf(mx, v[i]); }
  red[t] = mx;
  __syncthreads();
  for (int s = 128; s > 0; s >>= 1) {
    if (t < s) red[t] = fmaxf(red[t], red[t + s]);
    __syncthreads();
  }
  mx = red[0];
  __syncthreads();

  float sum = 0.f;
#pragma unroll
  for (int i = 0; i < 8; ++i) { v[i] = __expf(v[i] - mx); sum += v[i]; }
  red[t] = sum;
  __syncthreads();
  for (int s = 128; s > 0; s >>= 1) {
    if (t < s) red[t] += red[t + s];
    __syncthreads();
  }
  const float inv = 1.0f / red[0];
#pragma unroll
  for (int i = 0; i < 8; ++i) p[t + i * 256] = v[i] * inv;
}

// ---------------------------------------------------------------------------
// attn_out = attn_weights @ V per (b,h). A = fp32 attn (f16 fragments on the
// fly), B = Vt [B,H,D,S]. All 8 waves of a block share (b,h), so the 64x32
// Vt slab is TDM-staged in LDS (double-buffered). Output f16 flat [B,H,S,D]
// (== the reference's no-head-transpose reshape when read flat as [B*S,E]).
__global__ __launch_bounds__(256) void attnv_kernel(
    const float* __restrict__ attn, const f16* __restrict__ Vth, f16* __restrict__ AOh) {
  __shared__ __align__(16) f16 shB[2][64 * 32];
  const int lane = threadIdx.x & 31;
  const int wave = threadIdx.x >> 5;
  const int job  = blockIdx.x * 8 + wave;  // 8192 jobs: 64 bh x 128 mtile
  const int mtile = job & 127;
  const int bh    = job >> 7;              // uniform across the block
  const float* A  = attn + (size_t)bh * SS * SS;
  const f16*   Vt = Vth + (size_t)bh * DD * SS;   // 64 rows x SS, contiguous K
  const int m0 = mtile * 16;

  v8f acc[4] = {v8f{}, v8f{}, v8f{}, v8f{}};
#if HAVE_TDM
  if (wave == 0) {
    tdm_load_64x32(Vt, lds_off(&shB[0][0]), SS, 64, SS);
    __builtin_amdgcn_s_wait_tensorcnt(0);
  }
  __syncthreads();
  for (int kk = 0; kk < SS / 32; ++kk) {
    const int buf = kk & 1;
    if (wave == 0 && kk + 1 < SS / 32)
      tdm_load_64x32(Vt + (kk + 1) * 32, lds_off(&shB[buf ^ 1][0]), SS, 64, SS);
    v16h a = load_a_f32(A, SS, m0, kk * 32, lane);
#pragma unroll
    for (int j = 0; j < 4; ++j) {
      v16h b = load_b_lds(&shB[buf][0], j * 16, lane);
      acc[j] = WMMA_F16(a, b, acc[j]);
    }
    if (wave == 0 && kk + 1 < SS / 32) __builtin_amdgcn_s_wait_tensorcnt(0);
    __syncthreads();
  }
#else
  for (int kk = 0; kk < SS / 32; ++kk) {
    for (int idx = threadIdx.x; idx < 64 * 32; idx += 256)
      shB[0][idx] = Vt[(size_t)(idx >> 5) * SS + kk * 32 + (idx & 31)];
    __syncthreads();
    v16h a = load_a_f32(A, SS, m0, kk * 32, lane);
#pragma unroll
    for (int j = 0; j < 4; ++j) {
      v16h b = load_b_lds(&shB[0][0], j * 16, lane);
      acc[j] = WMMA_F16(a, b, acc[j]);
    }
    __syncthreads();
  }
#endif

  const int mrow = m0 + ((lane >> 4) << 3);
  const int ncol = lane & 15;
#pragma unroll
  for (int j = 0; j < 4; ++j) {
    const int d = j * 16 + ncol;
#pragma unroll
    for (int r = 0; r < 8; ++r) {
      AOh[((size_t)bh * SS + (mrow + r)) * DD + d] = (f16)acc[j][r];
    }
  }
}

// ---------------------------------------------------------------------------
// output = attn_out(flat [B*S, E]) @ Wo.T + bo, fp32 result to d_out.
// Same TDM-staged weight-slab scheme as qkv_proj.
__global__ __launch_bounds__(256) void oproj_kernel(
    const f16* __restrict__ AOh, const f16* __restrict__ Woh,
    const float* __restrict__ bo, float* __restrict__ out) {
  __shared__ __align__(16) f16 shB[2][64 * 32];
  const int lane = threadIdx.x & 31;
  const int wave = threadIdx.x >> 5;
  const int nquad = blockIdx.x & 15;                     // shared by the block
  const int m0    = (((int)blockIdx.x >> 4) * 8 + wave) * 16;
  const int n0    = nquad * 64;
  const f16* Bsrc = Woh + (size_t)n0 * EE;

  v8f acc[4] = {v8f{}, v8f{}, v8f{}, v8f{}};
#if HAVE_TDM
  if (wave == 0) {
    tdm_load_64x32(Bsrc, lds_off(&shB[0][0]), EE, 64, EE);
    __builtin_amdgcn_s_wait_tensorcnt(0);
  }
  __syncthreads();
  for (int kk = 0; kk < EE / 32; ++kk) {
    const int buf = kk & 1;
    if (wave == 0 && kk + 1 < EE / 32)
      tdm_load_64x32(Bsrc + (kk + 1) * 32, lds_off(&shB[buf ^ 1][0]), EE, 64, EE);
    v16h a = load_a_f16(AOh, EE, m0, kk * 32, lane);
#pragma unroll
    for (int j = 0; j < 4; ++j) {
      v16h b = load_b_lds(&shB[buf][0], j * 16, lane);
      acc[j] = WMMA_F16(a, b, acc[j]);
    }
    if (wave == 0 && kk + 1 < EE / 32) __builtin_amdgcn_s_wait_tensorcnt(0);
    __syncthreads();
  }
#else
  for (int kk = 0; kk < EE / 32; ++kk) {
    for (int idx = threadIdx.x; idx < 64 * 32; idx += 256)
      shB[0][idx] = Bsrc[(size_t)(idx >> 5) * EE + kk * 32 + (idx & 31)];
    __syncthreads();
    v16h a = load_a_f16(AOh, EE, m0, kk * 32, lane);
#pragma unroll
    for (int j = 0; j < 4; ++j) {
      v16h b = load_b_lds(&shB[0][0], j * 16, lane);
      acc[j] = WMMA_F16(a, b, acc[j]);
    }
    __syncthreads();
  }
#endif

  const int mrow = m0 + ((lane >> 4) << 3);
  const int ncol = lane & 15;
#pragma unroll
  for (int j = 0; j < 4; ++j) {
    const int n = n0 + j * 16 + ncol;
    const float bn = bo[n];
#pragma unroll
    for (int r = 0; r < 8; ++r) {
      out[(size_t)(mrow + r) * EE + n] = acc[j][r] + bn;
    }
  }
}

// ---------------------------------------------------------------------------
extern "C" void kernel_launch(void* const* d_in, const int* in_sizes, int n_in,
                              void* d_out, int out_size, void* d_ws, size_t ws_size,
                              hipStream_t stream) {
  (void)in_sizes; (void)n_in; (void)out_size; (void)ws_size;
  const float* x  = (const float*)d_in[0];
  const float* Wq = (const float*)d_in[1];
  const float* bq = (const float*)d_in[2];
  const float* Wk = (const float*)d_in[3];
  const float* bk = (const float*)d_in[4];
  const float* Wv = (const float*)d_in[5];
  const float* bv = (const float*)d_in[6];
  const float* Wo = (const float*)d_in[7];
  const float* bo = (const float*)d_in[8];

  float* out  = (float*)d_out;                       // [B,S,E]  = 8,388,608 f32
  float* attn = out + (size_t)BB * SS * EE;          // [B,H,S,S] = 268,435,456 f32

  // Workspace carve-up (f16 intermediates, ~92 MB total).
  char* ws = (char*)d_ws;
  size_t off = 0;
  auto carve = [&](size_t bytes) -> void* {
    void* p = ws + off;
    off += (bytes + 255) & ~(size_t)255;
    return p;
  };
  const size_t nBSE = (size_t)BB * SS * EE;
  f16* xh  = (f16*)carve(nBSE * 2);
  f16* Wqh = (f16*)carve((size_t)EE * EE * 2);
  f16* Wkh = (f16*)carve((size_t)EE * EE * 2);
  f16* Wvh = (f16*)carve((size_t)EE * EE * 2);
  f16* Woh = (f16*)carve((size_t)EE * EE * 2);
  f16* Qh  = (f16*)carve(nBSE * 2);
  f16* Kh  = (f16*)carve(nBSE * 2);
  f16* Vth = (f16*)carve(nBSE * 2);
  f16* AOh = (f16*)carve(nBSE * 2);

  // 1) fp32 -> f16 casts (x + 4 weight matrices)
  {
    int n4 = (int)(nBSE / 4);                           // 2,097,152
    cast4_kernel<<<n4 / 256, 256, 0, stream>>>(x, xh, n4);
    int w4 = (EE * EE) / 4;                             // 262,144
    cast4_kernel<<<w4 / 256, 256, 0, stream>>>(Wq, Wqh, w4);
    cast4_kernel<<<w4 / 256, 256, 0, stream>>>(Wk, Wkh, w4);
    cast4_kernel<<<w4 / 256, 256, 0, stream>>>(Wv, Wvh, w4);
    cast4_kernel<<<w4 / 256, 256, 0, stream>>>(Wo, Woh, w4);
  }

  // 2) QKV projections (fused, blockIdx.y selects Q/K/V)
  qkv_proj_kernel<<<dim3(1024, 3), 256, 0, stream>>>(xh, Wqh, Wkh, Wvh,
                                                     bq, bk, bv, Qh, Kh, Vth);

  // 3) scores = QK^T / 8 -> d_out attn region (fp32)
  scores_kernel<<<32768, 256, 0, stream>>>(Qh, Kh, attn);

  // 4) softmax over last dim, in place
  softmax_kernel<<<BB * HH * SS, 256, 0, stream>>>(attn);

  // 5) attn @ V -> f16 [B,H,S,D]
  attnv_kernel<<<1024, 256, 0, stream>>>(attn, Vth, AOh);

  // 6) output projection -> d_out (fp32)
  oproj_kernel<<<1024, 256, 0, stream>>>(AOh, Woh, bo, out);
}
